// SpectralConv2d_56710748177058
// MI455X (gfx1250) — matrix operand order repaired and checked by reading, not compiled, not run on hardware
//
#include <hip/hip_runtime.h>
#include <hip/hip_bf16.h>
#include <math.h>

// ---------------------------------------------------------------------------
// Spectral conv (FNO layer) as DFT-GEMMs on CDNA5 WMMA.
//   x:[16,32,256,256] f32 -> out:[16,32,256,256] f32
// Truncated rfft2/irfft2 (32 x-modes, 16 y-modes) expressed as dense DFT
// matmuls on v_wmma_f32_16x16x32_f16. All intermediates laid out so WMMA
// fragments are contiguous vector loads (no scalar gathers, no spills).
// ---------------------------------------------------------------------------

typedef __attribute__((ext_vector_type(16))) _Float16     v16h;
typedef __attribute__((ext_vector_type(8)))  _Float16     v8h;
typedef __attribute__((ext_vector_type(8)))  float        v8f;
typedef __attribute__((ext_vector_type(4)))  unsigned int v4u;
typedef __attribute__((ext_vector_type(8)))  int          v8i;
typedef __attribute__((ext_vector_type(4)))  int          v4i;

#define WMMA(A, B, Cacc) \
  __builtin_amdgcn_wmma_f32_16x16x32_f16(false, (A), false, (B), (short)0, (Cacc), false, false)

// TDM availability (device pass only; host pass falls back to plain copy)
#if defined(__has_builtin)
#if __has_builtin(__builtin_amdgcn_tensor_load_to_lds)
#define HAVE_TDM 1
#endif
#endif
#ifndef HAVE_TDM
#define HAVE_TDM 0
#endif

__device__ __host__ constexpr float kTwoPiOverN = 6.283185307179586f / 256.0f;

// Workspace layout (units: _Float16). All offsets 32B aligned.
constexpr size_t OFF_B1C  = 0;                    // 4096  : stage1 B,  cos
constexpr size_t OFF_B1S  = 4096;                 // 4096  : stage1 B, -sin
constexpr size_t OFF_A2C  = 8192;                 // 8192  : stage2 A,  cos
constexpr size_t OFF_A2SP = 16384;                // 8192  : stage2 A, +sin
constexpr size_t OFF_A2SN = 24576;                // 8192  : stage2 A, -sin
constexpr size_t OFF_A3C  = 32768;                // 8192  : stageC1 A, cos/256
constexpr size_t OFF_A3SP = 40960;                // 8192  : stageC1 A, +sin/256
constexpr size_t OFF_A3SN = 49152;                // 8192  : stageC1 A, -sin/256
constexpr size_t OFF_B4   = 57344;                // 8192  : stageC2 B (irfft-y)
constexpr size_t OFF_WR   = 65536;                // 524288: packed W real
constexpr size_t OFF_WI   = OFF_WR + 524288;      // 524288: packed W imag
constexpr size_t OFF_WIN  = OFF_WI + 524288;      // 524288: packed -W imag
constexpr size_t OFF_Z    = OFF_WIN + 524288;     // 524288: Z[p][kx][ky][b][i]
constexpr size_t OFF_OFT  = OFF_Z + 524288;       // 524288: Oft[p][b][o][ky][kx]
constexpr size_t ZPSTRIDE = 262144;               // p-plane stride (halves)

__device__ inline int kxmap(int r) { return (r < 16) ? r : r + 224; }  // 240..255

// ---------------------------------------------------------------------------
// Init: DFT twiddle tables pre-packed in WMMA fragment order.
//  B-frag half (t,lane,h): K = 32t + 16*(h>>3) + 8*(lane>=16) + (h&7), N = lane&15
//  A-frag half (t,lane,h): M = lane&15, K = 32t + 16*(lane>=16) + h
// ---------------------------------------------------------------------------
__global__ void fno_init_twiddles(_Float16* __restrict__ ws) {
  int idx = blockIdx.x * blockDim.x + threadIdx.x;
  if (idx < 4096) {  // stage1 B tables: [t=8][lane][h], K = y, N = ky
    int t = idx >> 9, lane = (idx >> 4) & 31, h = idx & 15;
    int y = 32 * t + 16 * (h >> 3) + 8 * (lane >> 4) + (h & 7);
    int n = lane & 15;
    float th = kTwoPiOverN * (float)((y * n) & 255);
    ws[OFF_B1C + idx] = (_Float16)cosf(th);
    ws[OFF_B1S + idx] = (_Float16)(-sinf(th));
  } else if (idx < 12288) {  // stage2 A tables: [s=2][t=8], M = kx-row, K = x
    int li = idx - 4096;
    int slot = li >> 9, lane = (li >> 4) & 31, h = li & 15;
    int s = slot >> 3, t = slot & 7;
    int row = 16 * s + (lane & 15);
    int x = 32 * t + 16 * (lane >> 4) + h;
    float th = kTwoPiOverN * (float)((kxmap(row) * x) & 255);
    ws[OFF_A2C + li]  = (_Float16)cosf(th);
    ws[OFF_A2SP + li] = (_Float16)sinf(th);
    ws[OFF_A2SN + li] = (_Float16)(-sinf(th));
  } else if (idx < 20480) {  // stageC1 A tables: [m=16], M = x, K = kx-row (32)
    int li = idx - 12288;
    int m = li >> 9, lane = (li >> 4) & 31, h = li & 15;
    int xr = 16 * m + (lane & 15);
    int kc = 16 * (lane >> 4) + h;
    float th = kTwoPiOverN * (float)((kxmap(kc) * xr) & 255);
    const float inv = 1.0f / 256.0f;  // 1/N from x-axis ifft
    ws[OFF_A3C + li]  = (_Float16)(cosf(th) * inv);
    ws[OFF_A3SP + li] = (_Float16)(sinf(th) * inv);
    ws[OFF_A3SN + li] = (_Float16)(-sinf(th) * inv);
  } else if (idx < 28672) {  // stageC2 B: [nt=16], K = kcat(32: Ur|Ui), N = y
    int li = idx - 20480;
    int nt = li >> 9, lane = (li >> 4) & 31, h = li & 15;
    int kcat = 16 * (h >> 3) + 8 * (lane >> 4) + (h & 7);
    int y = 16 * nt + (lane & 15);
    const float inv = 1.0f / 256.0f;  // 1/N from y-axis irfft
    float v;
    if (kcat < 16) {  // cos half: k==0 contributes once, k>0 twice (hermitian)
      v = (kcat == 0) ? inv : 2.0f * inv * cosf(kTwoPiOverN * (float)((kcat * y) & 255));
    } else {          // -sin half
      int k = kcat - 16;
      v = (k == 0) ? 0.0f : -2.0f * inv * sinf(kTwoPiOverN * (float)((k * y) & 255));
    }
    ws[OFF_B4 + li] = (_Float16)v;
  }
}

// Pack mixing weights into per-mode B fragments: K = in-ch (32), N = out-ch.
__global__ void fno_init_wpack(const float* __restrict__ w1r, const float* __restrict__ w1i,
                               const float* __restrict__ w2r, const float* __restrict__ w2i,
                               _Float16* __restrict__ ws) {
  int idx = blockIdx.x * blockDim.x + threadIdx.x;  // 0..524287
  int h = idx & 15, lane = (idx >> 4) & 31, nt = (idx >> 9) & 1, m = idx >> 10;
  int kxr = m >> 4, ky = m & 15;
  int i = 16 * (h >> 3) + 8 * (lane >> 4) + (h & 7);   // K: input channel
  int o = 16 * nt + (lane & 15);  // N: output channel
  const float* wr = (kxr < 16) ? w1r : w2r;
  const float* wi = (kxr < 16) ? w1i : w2i;
  int mx = kxr & 15;
  size_t src = (((size_t)i * 32 + o) * 16 + mx) * 16 + ky;  // w[i][o][mx][my]
  float r = wr[src], im = wi[src];
  ws[OFF_WR + idx]  = (_Float16)r;
  ws[OFF_WI + idx]  = (_Float16)im;
  ws[OFF_WIN + idx] = (_Float16)(-im);
}

// ---------------------------------------------------------------------------
// Kernel 1: forward truncated DFT. One block per (b,c) image, 8 waves.
//  LDS: image 256x256 f32 (TDM-loaded), Yt = [ky][x] f16 (transposed so
//  stage-2 B-frags are single 32B ds loads).
// ---------------------------------------------------------------------------
__global__ void fno_forward(const float* __restrict__ x, _Float16* __restrict__ ws) {
  extern __shared__ char smem[];
  float* As = (float*)smem;                      // 256*256 f32 = 256 KB
  _Float16* YtR = (_Float16*)(smem + 262144);    // [ky=16][x=256] f16
  _Float16* YtI = YtR + 4096;

  const int img = blockIdx.x;  // b*32 + c
  const int b = img >> 5, c = img & 31;
  const int tid = threadIdx.x, lane = tid & 31, wv = tid >> 5;
  const int laneHi = lane >> 4, laneLo = lane & 15;
  const float* src = x + (size_t)img * 65536;

#if HAVE_TDM
  if (wv == 0) {  // one TDM DMA of the whole 256x256 f32 tile into LDS
    unsigned long long ga = (unsigned long long)src;
    v4u g0;
    g0[0] = 1u;                                        // count=1, user mode
    g0[1] = 0u;                                        // lds_addr = 0
    g0[2] = (unsigned)(ga & 0xffffffffu);              // global_addr[31:0]
    g0[3] = (unsigned)((ga >> 32) & 0x01ffffffu) | (2u << 30);  // addr[56:32] | type=2
    v8i g1;
    g1[0] = (2 << 16);          // data_size = 4B
    g1[1] = (int)(256u << 16);  // tensor_dim0 = 256
    g1[2] = (int)(256u << 16);  // tensor_dim1 = 256
    g1[3] = (int)(256u << 16);  // tile_dim0 = 256
    g1[4] = 256;                // tile_dim1 = 256, tile_dim2 = 0
    g1[5] = 256;                // tensor_dim0_stride = 256
    g1[6] = 0;
    g1[7] = 0;
    v4i gz = {};
#if __clang_major__ >= 23
    v8i gz8 = {};
    __builtin_amdgcn_tensor_load_to_lds(g0, g1, gz, gz, gz8, 0);
#else
    __builtin_amdgcn_tensor_load_to_lds(g0, g1, gz, gz, 0);
#endif
    __builtin_amdgcn_s_wait_tensorcnt(0);
  }
  __syncthreads();
#else
  for (int i = tid; i < 65536 / 4; i += blockDim.x)
    ((float4*)As)[i] = ((const float4*)src)[i];
  __syncthreads();
#endif

  // Stage 1: Y[x,ky] = sum_y A[x,y] * e^{-i 2pi ky y/256}. 16 row-strips.
  const v16h* B1c = (const v16h*)(ws + OFF_B1C);
  const v16h* B1s = (const v16h*)(ws + OFF_B1S);
#pragma unroll 1
  for (int m = wv; m < 16; m += 8) {
    int row = 16 * m + laneLo;
    v8f cr = {}, ci = {};
#pragma unroll 1
    for (int t = 0; t < 8; ++t) {  // keep B-frags just-in-time (no spills)
      int y0 = 32 * t + 8 * laneHi;
      const float4* r0 = (const float4*)(As + row * 256 + y0);
      const float4* r1 = (const float4*)(As + row * 256 + y0 + 16);
      float4 f0 = r0[0], f1 = r0[1], f2 = r1[0], f3 = r1[1];
      v16h a;
      a[0] = (_Float16)f0.x;  a[1] = (_Float16)f0.y;
      a[2] = (_Float16)f0.z;  a[3] = (_Float16)f0.w;
      a[4] = (_Float16)f1.x;  a[5] = (_Float16)f1.y;
      a[6] = (_Float16)f1.z;  a[7] = (_Float16)f1.w;
      a[8]  = (_Float16)f2.x; a[9]  = (_Float16)f2.y;
      a[10] = (_Float16)f2.z; a[11] = (_Float16)f2.w;
      a[12] = (_Float16)f3.x; a[13] = (_Float16)f3.y;
      a[14] = (_Float16)f3.z; a[15] = (_Float16)f3.w;
      cr = WMMA(a, B1c[t * 32 + lane], cr);
      ci = WMMA(a, B1s[t * 32 + lane], ci);
    }
    // D tile -> Yt[ky][x]; 8 consecutive x per lane = one 16B ds store
    v8h yr8, yi8;
#pragma unroll
    for (int r = 0; r < 8; ++r) { yr8[r] = (_Float16)cr[r]; yi8[r] = (_Float16)ci[r]; }
    int xb = laneLo * 256 + 16 * m + 8 * laneHi;
    *(v8h*)(YtR + xb) = yr8;
    *(v8h*)(YtI + xb) = yi8;
  }
  __syncthreads();

  // Stage 2: Z[kx,ky] = sum_x e^{-i 2pi kx x/256} * Y[x,ky].
  // Zr = cos*Yr + sin*Yi ; Zi = cos*Yi - sin*Yr.  Jobs: (strip s, part p).
  const v16h* A2c  = (const v16h*)(ws + OFF_A2C);
  const v16h* A2sp = (const v16h*)(ws + OFF_A2SP);
  const v16h* A2sn = (const v16h*)(ws + OFF_A2SN);
  if (wv < 4) {
    int s = wv >> 1, p = wv & 1;
    v8f acc = {};
#pragma unroll 1
    for (int t = 0; t < 8; ++t) {
      int yb = laneLo * 256 + 32 * t + 16 * laneHi;  // [ky][x], 16 contiguous x
      v16h br = *(const v16h*)(YtR + yb);
      v16h bi = *(const v16h*)(YtI + yb);
      int fi = (s * 8 + t) * 32 + lane;
      if (p == 0) {
        acc = WMMA(A2c[fi], br, acc);
        acc = WMMA(A2sp[fi], bi, acc);
      } else {
        acc = WMMA(A2c[fi], bi, acc);
        acc = WMMA(A2sn[fi], br, acc);
      }
    }
    // Z[p][kx][ky][b][i]
    _Float16* Z = ws + OFF_Z;
#pragma unroll
    for (int r = 0; r < 8; ++r) {
      int kxr = 16 * s + (laneHi ? r + 8 : r);
      Z[((((size_t)p * 32 + kxr) * 16 + laneLo) * 16 + b) * 32 + c] = (_Float16)acc[r];
    }
  }
}

// ---------------------------------------------------------------------------
// Kernel 2: channel mixing. Per mode (kx,ky): O[b,o] = sum_i Z[b,i] * W[i,o]
// (complex via 4 real WMMAs). 8 modes per block (one per wave).
// ---------------------------------------------------------------------------
__global__ void fno_mix(_Float16* __restrict__ ws) {
  const int tid = threadIdx.x, lane = tid & 31, wv = tid >> 5;
  const int laneHi = lane >> 4, laneLo = lane & 15;
  const int mode = blockIdx.x * 8 + wv;  // 0..511
  const int kxr = mode >> 4, ky = mode & 15;
  const _Float16* Z = ws + OFF_Z;
  _Float16* Oft = ws + OFF_OFT;
  const v16h* Wr  = (const v16h*)(ws + OFF_WR);
  const v16h* Wi  = (const v16h*)(ws + OFF_WI);
  const v16h* Win = (const v16h*)(ws + OFF_WIN);

  // A-frags: M = batch (laneLo), K = in channel; Z[p][kx][ky][b][i], i contiguous
  size_t zb = (((size_t)kxr * 16 + ky) * 16 + laneLo) * 32 + 8 * laneHi;
  v8h zr0 = *(const v8h*)(Z + zb);
  v8h zr1 = *(const v8h*)(Z + zb + 16);
  v8h zi0 = *(const v8h*)(Z + ZPSTRIDE + zb);
  v8h zi1 = *(const v8h*)(Z + ZPSTRIDE + zb + 16);
  v16h zr, zi;
#pragma unroll
  for (int j = 0; j < 8; ++j) {
    zr[j] = zr0[j]; zr[j + 8] = zr1[j];
    zi[j] = zi0[j]; zi[j + 8] = zi1[j];
  }
#pragma unroll 1
  for (int nt = 0; nt < 2; ++nt) {
    size_t fi = ((size_t)mode * 2 + nt) * 32 + lane;
    v8f orr = {}, oii = {};
    orr = WMMA(zr, Wr[fi], orr);   // Zr*Wr
    orr = WMMA(zi, Win[fi], orr);  // - Zi*Wi
    oii = WMMA(zr, Wi[fi], oii);   // Zr*Wi
    oii = WMMA(zi, Wr[fi], oii);   // + Zi*Wr
    // Oft[p][b][o][ky][kx]
#pragma unroll
    for (int r = 0; r < 8; ++r) {
      int bb = laneHi ? r + 8 : r;
      int o = 16 * nt + laneLo;
      size_t ob = ((((size_t)bb * 32 + o) * 16 + ky) * 32) + kxr;
      Oft[ob]            = (_Float16)orr[r];
      Oft[ZPSTRIDE + ob] = (_Float16)oii[r];
    }
  }
}

// ---------------------------------------------------------------------------
// Kernel 3: inverse transform. One block per (b,o) image.
//  C1: U[x,ky] = (1/256) sum_kx e^{+i 2pi kx x/256} * Oft[kx,ky]  (to LDS)
//  C2: out[x,y] = sum_k Ur*Cy + Ui*Sy   (K=32 concat, 16 y-tiles)
// ---------------------------------------------------------------------------
__global__ void fno_inverse(const _Float16* __restrict__ ws, float* __restrict__ out) {
  extern __shared__ char smem[];
  _Float16* U = (_Float16*)smem;  // [x=256][32] f16 : [x][Ur(0..15)|Ui(16..31)]
  const int img = blockIdx.x;
  const int b = img >> 5, o = img & 31;
  const int tid = threadIdx.x, lane = tid & 31, wv = tid >> 5;
  const int laneHi = lane >> 4, laneLo = lane & 15;
  const _Float16* Oft = ws + OFF_OFT;
  const v16h* A3c  = (const v16h*)(ws + OFF_A3C);
  const v16h* A3sp = (const v16h*)(ws + OFF_A3SP);
  const v16h* A3sn = (const v16h*)(ws + OFF_A3SN);
  const v16h* B4   = (const v16h*)(ws + OFF_B4);

  // B-frags: K = kx (16*laneHi+h, contiguous), N = ky (laneLo); one 32B load
  size_t ob = (((size_t)b * 32 + o) * 16 + laneLo) * 32 + 16 * laneHi;
  v16h br = *(const v16h*)(Oft + ob);
  v16h bi = *(const v16h*)(Oft + ZPSTRIDE + ob);

#pragma unroll 1
  for (int m = wv; m < 16; m += 8) {
    v8f ur = {}, ui = {};
    ur = WMMA(A3c[m * 32 + lane], br, ur);   //  cos*Or
    ur = WMMA(A3sn[m * 32 + lane], bi, ur);  // -sin*Oi
    ui = WMMA(A3c[m * 32 + lane], bi, ui);   //  cos*Oi
    ui = WMMA(A3sp[m * 32 + lane], br, ui);  // +sin*Or
#pragma unroll
    for (int r = 0; r < 8; ++r) {
      int xr = 16 * m + (laneHi ? r + 8 : r);
      U[xr * 32 + laneLo]      = (_Float16)ur[r];
      U[xr * 32 + 16 + laneLo] = (_Float16)ui[r];
    }
  }
  __syncthreads();

  float* dst = out + (size_t)img * 65536;
#pragma unroll 1
  for (int j = wv; j < 256; j += 8) {  // 16 x-strips * 16 y-tiles
    int m = j >> 4, nt = j & 15;
    // A-frag from LDS U: M = x, K = kcat; two contiguous 16B runs per lane
    int base = (16 * m + laneLo) * 32 + 8 * laneHi;
    v8h a0 = *(const v8h*)(U + base);
    v8h a1 = *(const v8h*)(U + base + 16);
    v16h a;
#pragma unroll
    for (int h = 0; h < 8; ++h) { a[h] = a0[h]; a[h + 8] = a1[h]; }
    v8f acc = {};
    acc = WMMA(a, B4[nt * 32 + lane], acc);
#pragma unroll
    for (int r = 0; r < 8; ++r) {
      int xo = 16 * m + (laneHi ? r + 8 : r);
      dst[xo * 256 + 16 * nt + laneLo] = acc[r];
    }
  }
}

// ---------------------------------------------------------------------------
extern "C" void kernel_launch(void* const* d_in, const int* in_sizes, int n_in,
                              void* d_out, int out_size, void* d_ws, size_t ws_size,
                              hipStream_t stream) {
  (void)in_sizes; (void)n_in; (void)out_size; (void)ws_size;
  const float* x   = (const float*)d_in[0];
  const float* w1r = (const float*)d_in[1];
  const float* w1i = (const float*)d_in[2];
  const float* w2r = (const float*)d_in[3];
  const float* w2i = (const float*)d_in[4];
  _Float16* ws = (_Float16*)d_ws;
  float* out = (float*)d_out;

  fno_init_twiddles<<<112, 256, 0, stream>>>(ws);                     // 28672 slots
  fno_init_wpack<<<2048, 256, 0, stream>>>(w1r, w1i, w2r, w2i, ws);   // 524288 slots
  fno_forward<<<512, 256, 262144 + 16384, stream>>>(x, ws);           // 1 block / image
  fno_mix<<<64, 256, 0, stream>>>(ws);                                // 8 modes / block
  fno_inverse<<<512, 256, 256 * 32 * 2, stream>>>(ws, out);           // 1 block / image
}
